// KGRAMS_43559558316069
// MI455X (gfx1250) — matrix operand order repaired and checked by reading, not compiled, not run on hardware
//
#include <hip/hip_runtime.h>
#include <hip/hip_bf16.h>

typedef __attribute__((ext_vector_type(16))) _Float16 v16h;
typedef __attribute__((ext_vector_type(8)))  float    v8f;

#define V_    50000
#define E_    128
#define C_    100
#define F_    3
#define B_    256
#define R_    10
#define L_    80
#define D_    64
#define H_    128
#define LAT_  64
#define Z_    128          // D+LAT
#define KCONV 384          // F*E
#define NPAD  112          // C padded to 7 WMMA tiles
#define MPOS  78           // L-F+1

// ---------------- workspace byte offsets (all 256B aligned) ----------------
#define WS_BU      0u        // 112*384 f16   = 86016
#define WS_BI      86016u    // 112*384 f16
#define WS_WCAT    172032u   // 512*256 f16   = 262144
#define WS_HF16    434176u   // 256*128 f16   = 65536
#define WS_FEATSU  499712u   // 2560*100 f32  = 1024000
#define WS_FEATSI  1523712u
#define WS_UF      2547712u  // 256*128 f32
#define WS_ITF     2678784u
#define WS_G       2809856u  // 256*512 f32
#define WS_H       3334144u  // 256*128 f32
#define WS_C       3465216u  // 256*128 f32

// ---------------------------------------------------------------------------
// 0) One-time weight conversion: conv weights -> padded f16 B-matrices,
//    [W_ih | W_hh] -> concatenated f16 [512][256]
// ---------------------------------------------------------------------------
__global__ __launch_bounds__(256) void prep_kernel(
    const float* __restrict__ u_convw, const float* __restrict__ i_convw,
    const float* __restrict__ W_ih,    const float* __restrict__ W_hh,
    _Float16* __restrict__ Bu, _Float16* __restrict__ Bi,
    _Float16* __restrict__ Wcat)
{
  int idx = blockIdx.x * 256 + threadIdx.x;
  if (idx < NPAD * KCONV) {
    int n = idx / KCONV, k = idx % KCONV;
    Bu[idx] = (_Float16)((n < C_) ? u_convw[n * KCONV + k] : 0.0f);
    Bi[idx] = (_Float16)((n < C_) ? i_convw[n * KCONV + k] : 0.0f);
  }
  if (idx < 512 * 256) {
    int n = idx >> 8, k = idx & 255;
    Wcat[idx] = (_Float16)((k < E_) ? W_ih[n * E_ + k] : W_hh[n * H_ + (k - E_)]);
  }
}

// ---------------------------------------------------------------------------
// 1) Text-CNN + ReLU + max-pool per review (WMMA GEMM: M=78pad80, N=100pad112,
//    K=384).  One workgroup (4 waves) per (entity, b, r).
//    Epilogue is branchless: padded rows/channels contribute 0 to a max whose
//    initial value is 0 (== relu floor), so ds_max can run unconditionally.
// ---------------------------------------------------------------------------
__global__ __launch_bounds__(128) void conv_kernel(
    const int* __restrict__ u_reviews, const int* __restrict__ i_reviews,
    const float* __restrict__ word_emb,
    const _Float16* __restrict__ BuF16, const _Float16* __restrict__ BiF16,
    const float* __restrict__ u_convb, const float* __restrict__ i_convb,
    float* __restrict__ feats_u, float* __restrict__ feats_i)
{
  __shared__ _Float16 sEmb[82][136];   // rows 80,81 zero pad (patch overhang)
  __shared__ int      sW[82];
  __shared__ unsigned sMax[NPAD];
  __shared__ float    sBias[NPAD];

  int gid = blockIdx.x;
  int ent = gid / (B_ * R_);
  int rid = gid % (B_ * R_);
  const int*      rv    = (ent == 0 ? u_reviews : i_reviews) + rid * L_;
  const _Float16* Wb    = (ent == 0 ? BuF16 : BiF16);
  const float*    cb    = (ent == 0 ? u_convb : i_convb);
  float*          feats = (ent == 0 ? feats_u : feats_i) + rid * C_;

  int tid = threadIdx.x;
  if (tid < 82) sW[tid] = (tid < L_) ? rv[tid] : 0;
  __syncthreads();
  for (int idx = tid; idx < 82 * E_; idx += 128) {
    int row = idx >> 7, col = idx & 127;
    float v = (row < L_) ? word_emb[sW[row] * E_ + col] : 0.0f;
    sEmb[row][col] = (_Float16)v;
  }
  if (tid < NPAD) {
    sMax[tid]  = 0u;                       // 0u == 0.0f; relu(x) = max(0,x)
    sBias[tid] = (tid < C_) ? cb[tid] : 0.0f;
  }
  __syncthreads();

  int wave = tid >> 5, lane = tid & 31;
  int half = lane >> 4, lm = lane & 15;

  for (int tp = wave; tp < 35; tp += 4) {   // 5 M-tiles x 7 N-tiles
    int mt = tp / 7, nt = tp % 7;
    v8f acc = {};
    for (int ks = 0; ks < 12; ++ks) {       // K = 384 = 12 * 32
      int kbase = ks * 32;
      v16h a, bf;
      // A fragment: 16-bit A 16x32 layout (lanes 0-15 = M, half-wave K split)
      #pragma unroll
      for (int j = 0; j < 8; ++j) {
        int kk  = kbase + ((j < 4) ? 0 : 16) + half * 8 + 2 * (j & 3);
        int row = mt * 16 + lm + (kk >> 7);  // patch row + filter tap
        int e   = kk & 127;                  // kk even -> e,e+1 same tap
        a[2 * j]     = sEmb[row][e];
        a[2 * j + 1] = sEmb[row][e + 1];
      }
      // B fragment: 32x16, lane n = lm, half-wave K split
      const _Float16* wrow = Wb + (nt * 16 + lm) * KCONV;
      #pragma unroll
      for (int v = 0; v < 8; ++v) {
        int kk = kbase + ((v < 4) ? 0 : 16) + half * 8 + 2 * (v & 3);
        bf[2 * v]     = wrow[kk];
        bf[2 * v + 1] = wrow[kk + 1];
      }
      acc = __builtin_amdgcn_wmma_f32_16x16x32_f16(
          false, a, false, bf, (short)0, acc, false, false);
    }
    int chan = nt * 16 + lm;
    bool chanOK = (chan < C_);
    float bias = sBias[chan];
    #pragma unroll
    for (int v = 0; v < 8; ++v) {
      int m = mt * 16 + v + half * 8;
      float val = acc[v] + bias;
      val = (val > 0.0f) ? val : 0.0f;                   // relu
      val = (m < MPOS && chanOK) ? val : 0.0f;           // mask pads (no-op max)
      atomicMax(&sMax[chan], __float_as_uint(val));      // branchless ds_max
    }
  }
  __syncthreads();
  if (tid < C_) feats[tid] = __uint_as_float(sMax[tid]);
}

// ---------------------------------------------------------------------------
// 2) Attention + latent feature + id embedding.  One block per b.
// ---------------------------------------------------------------------------
__global__ __launch_bounds__(128) void attn_kernel(
    const float* __restrict__ feats,       // [B,R,C]
    const int* __restrict__ target_ids,    // [B]
    const int* __restrict__ score_ids,     // [B,R]
    const float* __restrict__ score_emb, const float* __restrict__ id_emb,
    const float* __restrict__ WO, const float* __restrict__ Wu,
    const float* __restrict__ b1, const float* __restrict__ b2,
    const float* __restrict__ hvec,
    const float* __restrict__ linw, const float* __restrict__ linb,
    float* __restrict__ outf)              // [B,Z]
{
  int b = blockIdx.x, tid = threadIdx.x;
  __shared__ float rf[R_][C_];
  __shared__ int   sids[R_];
  __shared__ float contrib[H_][R_];
  __shared__ float att[R_];
  __shared__ float imp[C_];

  if (tid < R_) sids[tid] = score_ids[b * R_ + tid];
  for (int idx = tid; idx < R_ * C_; idx += 128)
    rf[idx / C_][idx % C_] = feats[b * R_ * C_ + idx];
  __syncthreads();

  {   // thread tid == hidden unit h
    int h = tid;
    float hv = hvec[h];
    float accs[R_];
    #pragma unroll
    for (int r = 0; r < R_; ++r) accs[r] = b1[h];
    for (int c = 0; c < C_; ++c) {
      float w = WO[h * C_ + c];
      #pragma unroll
      for (int r = 0; r < R_; ++r) accs[r] += w * rf[r][c];
    }
    // faithful to the torch reshape [R,D]->[D,R]: se[d][r] = flat[d*R+r]
    for (int d = 0; d < D_; ++d) {
      float w = Wu[h * D_ + d];
      #pragma unroll
      for (int r = 0; r < R_; ++r) {
        int i = d * R_ + r;
        float se = score_emb[sids[i >> 6] * D_ + (i & 63)];
        accs[r] += w * se;
      }
    }
    #pragma unroll
    for (int r = 0; r < R_; ++r) {
      float a = accs[r] > 0.0f ? accs[r] : 0.0f;   // relu
      contrib[h][r] = hv * a;
    }
  }
  __syncthreads();
  if (tid == 0) {   // deterministic serial softmax over R=10
    float logits[R_]; float mx = -1e30f;
    for (int r = 0; r < R_; ++r) {
      float s = b2[0];
      for (int h = 0; h < H_; ++h) s += contrib[h][r];
      logits[r] = s; if (s > mx) mx = s;
    }
    float den = 0.0f;
    for (int r = 0; r < R_; ++r) { logits[r] = expf(logits[r] - mx); den += logits[r]; }
    for (int r = 0; r < R_; ++r) att[r] = logits[r] / den;
  }
  __syncthreads();
  if (tid < C_) {
    float s = 0.0f;
    for (int r = 0; r < R_; ++r) s += att[r] * rf[r][tid];
    imp[tid] = s;
  }
  __syncthreads();
  if (tid < LAT_) {
    float s = linb[tid];
    for (int c = 0; c < C_; ++c) s += linw[tid * C_ + c] * imp[c];
    outf[b * Z_ + D_ + tid] = s;
  }
  if (tid < D_)
    outf[b * Z_ + tid] = id_emb[target_ids[b] * D_ + tid];
}

// ---------------------------------------------------------------------------
// 3) Rating prediction + MSE loss (one block, deterministic reduction)
// ---------------------------------------------------------------------------
__global__ __launch_bounds__(256) void pred_kernel(
    const float* __restrict__ uf, const float* __restrict__ itf,
    const float* __restrict__ W1, const float* __restrict__ b_u,
    const float* __restrict__ b_i, const float* __restrict__ mu,
    const float* __restrict__ target, float* __restrict__ out)
{
  int b = threadIdx.x;
  __shared__ float sq[B_];
  float p = b_u[b] + b_i[b] + mu[b];
  for (int z = 0; z < Z_; ++z) p += uf[b * Z_ + z] * itf[b * Z_ + z] * W1[z];
  out[b] = p;
  float d = p - target[b]; sq[b] = d * d;
  __syncthreads();
  if (b == 0) {
    float s = 0.0f;
    for (int j = 0; j < B_; ++j) s += sq[j];
    out[B_] = s / (float)B_;
  }
}

// ---------------------------------------------------------------------------
// 4) LSTM initial state: c0/h0 = tanh(z1 @ W.T + b)
// ---------------------------------------------------------------------------
__global__ __launch_bounds__(128) void init_kernel(
    const float* __restrict__ uf, const float* __restrict__ itf,
    const float* __restrict__ c0w, const float* __restrict__ c0b,
    const float* __restrict__ h0w, const float* __restrict__ h0b,
    float* __restrict__ hState, float* __restrict__ cState,
    _Float16* __restrict__ hF16)
{
  int b = blockIdx.x, h = threadIdx.x;
  __shared__ float z1[2 * Z_];
  for (int i = h; i < Z_; i += 128) { z1[i] = uf[b * Z_ + i]; z1[Z_ + i] = itf[b * Z_ + i]; }
  __syncthreads();
  float sc = c0b[h], sh = h0b[h];
  for (int z = 0; z < 2 * Z_; ++z) {
    sc += z1[z] * c0w[h * 2 * Z_ + z];
    sh += z1[z] * h0w[h * 2 * Z_ + z];
  }
  float c0 = tanhf(sc), h0 = tanhf(sh);
  cState[b * H_ + h] = c0;
  hState[b * H_ + h] = h0;
  hF16[b * H_ + h] = (_Float16)h0;
}

// ---------------------------------------------------------------------------
// 5a) LSTM gate GEMM for one step: g = [x_t, h] @ [W_ih|W_hh].T + b
//     M=256 (16 blocks of one M-tile), N=512 (4 waves x 8 N-tiles), K=256
// ---------------------------------------------------------------------------
__global__ __launch_bounds__(128) void lstm_gemm_kernel(
    const int* __restrict__ target_reviews, int t,
    const float* __restrict__ word_emb, const _Float16* __restrict__ hF16,
    const _Float16* __restrict__ Wcat,
    const float* __restrict__ b_ih, const float* __restrict__ b_hh,
    float* __restrict__ g)
{
  __shared__ _Float16 sA[16][264];
  __shared__ int      sTW[16];
  int mt = blockIdx.x, tid = threadIdx.x;
  if (tid < 16) sTW[tid] = target_reviews[(mt * 16 + tid) * L_ + t];
  __syncthreads();
  for (int idx = tid; idx < 16 * 256; idx += 128) {
    int row = idx >> 8, col = idx & 255;
    int b = mt * 16 + row;
    _Float16 v = (col < E_) ? (_Float16)word_emb[sTW[row] * E_ + col]
                            : hF16[b * H_ + (col - E_)];
    sA[row][col] = v;
  }
  __syncthreads();

  int wave = tid >> 5, lane = tid & 31, half = lane >> 4, lm = lane & 15;
  for (int nti = 0; nti < 8; ++nti) {
    int nt = wave * 8 + nti;
    v8f acc = {};
    for (int ks = 0; ks < 8; ++ks) {        // K = 256 = 8 * 32
      int kbase = ks * 32;
      v16h a, bf;
      #pragma unroll
      for (int j = 0; j < 8; ++j) {
        int kk = kbase + ((j < 4) ? 0 : 16) + half * 8 + 2 * (j & 3);
        a[2 * j]     = sA[lm][kk];
        a[2 * j + 1] = sA[lm][kk + 1];
      }
      const _Float16* wrow = Wcat + (nt * 16 + lm) * 256;
      #pragma unroll
      for (int v = 0; v < 8; ++v) {
        int kk = kbase + ((v < 4) ? 0 : 16) + half * 8 + 2 * (v & 3);
        bf[2 * v]     = wrow[kk];
        bf[2 * v + 1] = wrow[kk + 1];
      }
      acc = __builtin_amdgcn_wmma_f32_16x16x32_f16(
          false, a, false, bf, (short)0, acc, false, false);
    }
    int n = nt * 16 + lm;
    float bias = b_ih[n] + b_hh[n];
    #pragma unroll
    for (int v = 0; v < 8; ++v) {
      int b = mt * 16 + v + half * 8;
      g[b * 512 + n] = acc[v] + bias;
    }
  }
}

// ---------------------------------------------------------------------------
// 5b) LSTM elementwise gate update (torch gate order i,f,g,o)
// ---------------------------------------------------------------------------
__global__ __launch_bounds__(256) void lstm_update_kernel(
    const float* __restrict__ g, float* __restrict__ hState,
    float* __restrict__ cState, _Float16* __restrict__ hF16,
    float* __restrict__ out_h, float* __restrict__ out_c)  // non-null on last step
{
  int idx = blockIdx.x * 256 + threadIdx.x;  // 0..B*H-1
  int b = idx >> 7, h = idx & 127;
  const float* gr = g + b * 512;
  float gi = gr[h], gf = gr[H_ + h], gg = gr[2 * H_ + h], go = gr[3 * H_ + h];
  float si = 1.0f / (1.0f + expf(-gi));
  float sf = 1.0f / (1.0f + expf(-gf));
  float so = 1.0f / (1.0f + expf(-go));
  float c = sf * cState[idx] + si * tanhf(gg);
  float hh = so * tanhf(c);
  cState[idx] = c; hState[idx] = hh; hF16[idx] = (_Float16)hh;
  if (out_h) { out_h[idx] = hh; out_c[idx] = c; }
}

// ---------------------------------------------------------------------------
extern "C" void kernel_launch(void* const* d_in, const int* in_sizes, int n_in,
                              void* d_out, int out_size, void* d_ws, size_t ws_size,
                              hipStream_t stream) {
  (void)in_sizes; (void)n_in; (void)out_size; (void)ws_size;
  // params pytree in dict-insertion order, then the 8 data inputs
  const float* word_emb = (const float*)d_in[0];
  const float* u_score_emb = (const float*)d_in[1];
  const float* u_id_emb    = (const float*)d_in[2];
  const float* u_convw     = (const float*)d_in[3];
  const float* u_convb     = (const float*)d_in[4];
  const float* u_WO        = (const float*)d_in[5];
  const float* u_Wu        = (const float*)d_in[6];
  const float* u_b1        = (const float*)d_in[7];
  const float* u_b2        = (const float*)d_in[8];
  const float* u_h         = (const float*)d_in[9];
  const float* u_linw      = (const float*)d_in[10];
  const float* u_linb      = (const float*)d_in[11];
  const float* i_score_emb = (const float*)d_in[12];
  const float* i_id_emb    = (const float*)d_in[13];
  const float* i_convw     = (const float*)d_in[14];
  const float* i_convb     = (const float*)d_in[15];
  const float* i_WO        = (const float*)d_in[16];
  const float* i_Wu        = (const float*)d_in[17];
  const float* i_b1        = (const float*)d_in[18];
  const float* i_b2        = (const float*)d_in[19];
  const float* i_h         = (const float*)d_in[20];
  const float* i_linw      = (const float*)d_in[21];
  const float* i_linb      = (const float*)d_in[22];
  const float* W1          = (const float*)d_in[23];
  const float* b_u         = (const float*)d_in[24];
  const float* b_i         = (const float*)d_in[25];
  const float* mu          = (const float*)d_in[26];
  const float* c0w         = (const float*)d_in[27];
  const float* c0b         = (const float*)d_in[28];
  const float* h0w         = (const float*)d_in[29];
  const float* h0b         = (const float*)d_in[30];
  const float* W_ih        = (const float*)d_in[31];
  const float* W_hh        = (const float*)d_in[32];
  const float* b_ih        = (const float*)d_in[33];
  const float* b_hh        = (const float*)d_in[34];
  const int* user_ids            = (const int*)d_in[35];
  const int* user_reviews        = (const int*)d_in[36];
  const int* item_ids_of_reviews = (const int*)d_in[37];
  const int* item_ids            = (const int*)d_in[38];
  const int* item_reviews        = (const int*)d_in[39];
  const int* user_ids_of_reviews = (const int*)d_in[40];
  const float* target_ratings    = (const float*)d_in[41];
  const int* target_reviews      = (const int*)d_in[42];

  float* out = (float*)d_out;
  char* ws = (char*)d_ws;
  _Float16* BuF16  = (_Float16*)(ws + WS_BU);
  _Float16* BiF16  = (_Float16*)(ws + WS_BI);
  _Float16* Wcat   = (_Float16*)(ws + WS_WCAT);
  _Float16* hF16   = (_Float16*)(ws + WS_HF16);
  float* feats_u   = (float*)(ws + WS_FEATSU);
  float* feats_i   = (float*)(ws + WS_FEATSI);
  float* uf        = (float*)(ws + WS_UF);
  float* itf       = (float*)(ws + WS_ITF);
  float* g         = (float*)(ws + WS_G);
  float* hState    = (float*)(ws + WS_H);
  float* cState    = (float*)(ws + WS_C);

  prep_kernel<<<512, 256, 0, stream>>>(u_convw, i_convw, W_ih, W_hh,
                                       BuF16, BiF16, Wcat);
  conv_kernel<<<2 * B_ * R_, 128, 0, stream>>>(
      user_reviews, item_reviews, word_emb, BuF16, BiF16,
      u_convb, i_convb, feats_u, feats_i);
  attn_kernel<<<B_, 128, 0, stream>>>(
      feats_u, user_ids, item_ids_of_reviews, u_score_emb, u_id_emb,
      u_WO, u_Wu, u_b1, u_b2, u_h, u_linw, u_linb, uf);
  attn_kernel<<<B_, 128, 0, stream>>>(
      feats_i, item_ids, user_ids_of_reviews, i_score_emb, i_id_emb,
      i_WO, i_Wu, i_b1, i_b2, i_h, i_linw, i_linb, itf);
  pred_kernel<<<1, 256, 0, stream>>>(uf, itf, W1, b_u, b_i, mu,
                                     target_ratings, out);
  init_kernel<<<B_, 128, 0, stream>>>(uf, itf, c0w, c0b, h0w, h0b,
                                      hState, cState, hF16);
  for (int t = 0; t < L_; ++t) {
    lstm_gemm_kernel<<<16, 128, 0, stream>>>(
        target_reviews, t, word_emb, hF16, Wcat, b_ih, b_hh, g);
    bool last = (t == L_ - 1);
    lstm_update_kernel<<<(B_ * H_) / 256, 256, 0, stream>>>(
        g, hState, cState, hF16,
        last ? (out + B_ + 1) : nullptr,
        last ? (out + B_ + 1 + B_ * H_) : nullptr);
  }
}
